// ItttLinear_19069654794325
// MI455X (gfx1250) — matrix operand order repaired and checked by reading, not compiled, not run on hardware
//
#include <hip/hip_runtime.h>
#include <hip/hip_bf16.h>
#include <math.h>

// Problem constants (from the reference)
#define BB    32
#define SS    64
#define DIN   2048
#define DOUT  2048
#define LR_SCALE       0.01f                 // 0.01*sqrt(2048)*sqrt(1/2048)
#define SCALAR_SCALER  45.254833995939045f   // sqrt(2048)

typedef __attribute__((ext_vector_type(16))) _Float16 v16h;
typedef __attribute__((ext_vector_type(8)))  float    v8f;

// ---------------------------------------------------------------------------
// Kernel 1: lr_eff[o,i] = LR_SCALE * exp(log_lr[o,i] * sqrt(DIN))
// 4M elements -> 16 MB in d_ws; amortizes the transcendental across 32 batches.
// ---------------------------------------------------------------------------
__global__ __launch_bounds__(256) void lr_precompute_kernel(
    const float* __restrict__ log_lr, float* __restrict__ lr_eff, int n)
{
    int i = blockIdx.x * blockDim.x + threadIdx.x;
    if (i < n) lr_eff[i] = LR_SCALE * __expf(log_lr[i] * SCALAR_SCALER);
}

// ---------------------------------------------------------------------------
// Kernel 2: Y[b,s,o] = sum_i x[b,s,i] * (lr_eff[o,i]*state[b,o,i] + base_w[o,i])
//
// Grid: (DOUT/128, B).  Block: 256 threads = 8 waves (wave32).
// Wave w owns the DISJOINT 16-wide output-column group cbase = obase + 16*w
// and all S=64 rows as four 16x16 WMMA tiles (4 accumulators along M).
//   -> each state element is loaded exactly once device-wide (512 MB stream),
//      as 128B-coalesced per-lane b32 loads with a fused lr*state+base FMA,
//      converted f32->f16 and fed to v_wmma_f32_16x16x32_f16 (f32 accum).
//   -> the x tile (64x32 f32) is re-read by all 8 waves, but x is 16 MB total
//      and L0/L2 resident, so that redundancy costs nothing at the HBM level.
// ---------------------------------------------------------------------------
__global__ __launch_bounds__(256) void ittt_linear_kernel(
    const float* __restrict__ x,       // [B, S, DIN]
    const float* __restrict__ lr,      // [DOUT, DIN] (precomputed lr_eff)
    const float* __restrict__ state,   // [B, DOUT, DIN]
    const float* __restrict__ base_w,  // [DOUT, DIN]
    float* __restrict__ out)           // [B, S, DOUT]
{
    const int batch = blockIdx.y;
    const int obase = blockIdx.x * 128;

    const int w    = threadIdx.x >> 5;   // wave id 0..7
    const int lane = threadIdx.x & 31;

    const int cbase = obase + 16 * w;    // this wave's 16 output columns

    // ---- A-fragment addressing (16-bit A 16x32 layout) ----
    // lane m in 0..15 : K segments [khalf, khalf+8) and [16+khalf, 16+khalf+8)
    const int m     = lane & 15;
    const int khalf = (lane >> 4) * 8;
    const float* px = x + ((size_t)(batch * SS + m) * DIN) + khalf;

    const size_t state_base = (size_t)batch * DOUT * DIN;

    v8f acc[4];   // 4 M-strips (S rows 0..15, 16..31, 32..47, 48..63)
    #pragma unroll
    for (int s = 0; s < 4; ++s) acc[s] = (v8f){};

    for (int kc = 0; kc < DIN; kc += 32) {
        // ------------- B fragment: W_eff tile, fused FMA, built ONCE -------
        // 16-bit B 32x16 layout: N = lane&15, element e -> K = khalf+e (e<8), 16+khalf+e-8 (e>=8).
        // Each lane reads runs of 8 consecutive K along one output row
        // load per array.  state is stream-once: non-temporal to keep L2 for
        // lr/base_w/x which are reused across all 32 batches.
        const int klane = kc + khalf;
        v16h bfrag;
        #pragma unroll
        for (int n = 0; n < 16; ++n) {
            const size_t off = (size_t)(cbase + m) * DIN + klane + (n < 8 ? n : n + 8);
            const float st = __builtin_nontemporal_load(&state[state_base + off]);
            const float wv = fmaf(lr[off], st, base_w[off]);
            bfrag[n] = (_Float16)wv;
        }

        // ------------- A fragments (4 row strips) + WMMA -------------------
        #pragma unroll
        for (int s = 0; s < 4; ++s) {
            const float* p = px + (size_t)(s * 16) * DIN + kc;
            v16h a;
            #pragma unroll
            for (int e = 0; e < 8; ++e) a[e]     = (_Float16)p[e];
            #pragma unroll
            for (int e = 0; e < 8; ++e) a[8 + e] = (_Float16)p[16 + e];

            acc[s] = __builtin_amdgcn_wmma_f32_16x16x32_f16(
                /*neg_a=*/false, a, /*neg_b=*/false, bfrag,
                /*c_mod=*/(short)0, acc[s],
                /*reuse_a=*/false, /*reuse_b=*/false);
        }
    }

    // ---------------- Store D (16x16 f32 C/D layout) --------------------
    // lanes 0-15: M = r,  N = lane;   lanes 16-31: M = 8 + r, N = lane-16
    const int nidx  = lane & 15;
    const int mhalf = (lane >> 4) * 8;
    #pragma unroll
    for (int s = 0; s < 4; ++s) {
        #pragma unroll
        for (int r = 0; r < 8; ++r) {
            out[(size_t)(batch * SS + s * 16 + mhalf + r) * DOUT
                + cbase + nidx] = acc[s][r];
        }
    }
}

// ---------------------------------------------------------------------------
// Launch
// Inputs (setup_inputs order): x, log_lr, state, momentum (unused), base_w
// ---------------------------------------------------------------------------
extern "C" void kernel_launch(void* const* d_in, const int* in_sizes, int n_in,
                              void* d_out, int out_size, void* d_ws, size_t ws_size,
                              hipStream_t stream)
{
    const float* x      = (const float*)d_in[0];
    const float* log_lr = (const float*)d_in[1];
    const float* state  = (const float*)d_in[2];
    // d_in[3] = momentum (zeros, unused by forward)
    const float* base_w = (const float*)d_in[4];
    float*       out    = (float*)d_out;
    float*       lr_eff = (float*)d_ws;   // 2048*2048 floats = 16 MB scratch

    const int n_lr = DOUT * DIN;
    lr_precompute_kernel<<<(n_lr + 255) / 256, 256, 0, stream>>>(log_lr, lr_eff, n_lr);

    dim3 grid(DOUT / 128, BB);
    ittt_linear_kernel<<<grid, 256, 0, stream>>>(x, lr_eff, state, base_w, out);
}
